// LSHSelfAttention_23081154249296
// MI455X (gfx1250) — matrix operand order repaired and checked by reading, not compiled, not run on hardware
//
#include <hip/hip_runtime.h>
#include <hip/hip_bf16.h>

// ---- config (matches reference) ----
// B=2, S=4096, HIDDEN=1024, H=16, D=64, NH=2, NB=128, CHUNK=64, n_chunks=128
// BH = B*H = 32. Token rows M = B*S = 8192.

typedef _Float16 v16h __attribute__((ext_vector_type(16)));
typedef float    v8f  __attribute__((ext_vector_type(8)));
typedef float    f4v  __attribute__((ext_vector_type(4)));
typedef _Float16 h8   __attribute__((ext_vector_type(8)));

#define WMMA_F16(a, b, c) \
  __builtin_amdgcn_wmma_f32_16x16x32_f16(false, (a), false, (b), (short)0, (c), false, false)

// ---- CDNA5 async copy to LDS (CDNA5 ISA, GVS mode) ----
// LDS address taken from a VGPR; global address = SGPR base + VGPR offset. 16B per lane per issue.
__device__ __forceinline__ unsigned lds_lo32(const void* p) {
  // flat LDS address: high 32 = shared aperture, low 32 = in-allocation offset
  return (unsigned)(unsigned long long)p;
}
__device__ __forceinline__ void async_ld16(unsigned lds_off, unsigned goff,
                                           const void* base) {
  asm volatile("global_load_async_to_lds_b128 %0, %1, %2"
               :: "v"(lds_off), "v"(goff),
                  "s"((unsigned long long)base)
               : "memory");
}
__device__ __forceinline__ void async_wait0() {
  asm volatile("s_wait_asynccnt 0x0" ::: "memory");
}

// ------------------------------------------------------------------
// K0: f32 -> f16 conversion pass (X, Wqk, Wv). 8 elems/thread, b128 I/O.
// ------------------------------------------------------------------
__global__ __launch_bounds__(256) void k_cvt(const float* __restrict__ src,
                                             _Float16* __restrict__ dst) {
  size_t i = ((size_t)blockIdx.x * 256 + threadIdx.x) * 8;
  f4v a = *(const f4v*)(src + i);
  f4v b = *(const f4v*)(src + i + 4);
  h8 o;
  o[0] = (_Float16)a[0]; o[1] = (_Float16)a[1];
  o[2] = (_Float16)a[2]; o[3] = (_Float16)a[3];
  o[4] = (_Float16)b[0]; o[5] = (_Float16)b[1];
  o[6] = (_Float16)b[2]; o[7] = (_Float16)b[3];
  *(h8*)(dst + i) = o;
}

// ------------------------------------------------------------------
// K1: projections qk = X @ Wqk^T, v = X @ Wv^T (f16 inputs via async-LDS).
// z==0 writes qk (f32 + f16 copies); z==1 writes v (f16 only).
// block 256 (8 waves), 64x64 tile, K-step 32, WMMA f16 -> f32 acc.
// ------------------------------------------------------------------
__global__ __launch_bounds__(256) void k_proj(const _Float16* __restrict__ Xh,
                                              const _Float16* __restrict__ Wh,
                                              float* __restrict__ qkb,
                                              _Float16* __restrict__ qkh,
                                              _Float16* __restrict__ vh) {
  __shared__ __attribute__((aligned(16))) _Float16 lsA[64 * 32];
  __shared__ __attribute__((aligned(16))) _Float16 lsB[64 * 32];
  const _Float16* Whz = Wh + (blockIdx.z ? 1048576 : 0);
  const int m0 = blockIdx.x * 64;
  const int n0 = blockIdx.y * 64;
  const int t = threadIdx.x, lane = t & 31, w = t >> 5;
  const int mi = w >> 1, ni0 = (w & 1) * 2;      // 16 subtiles over 8 waves
  v8f acc0 = {}, acc1 = {};

  const int Ml  = lane & 15;
  const int kbA = (lane < 16) ? 0 : 16;
  const int kbB = (lane < 16) ? 0 : 16;
  const int i0 = t * 8;                 // 8 halves = 16B per lane per tile
  const int rT = i0 >> 5, kT = i0 & 31; // staging row / k within tile

  for (int k0 = 0; k0 < 1024; k0 += 32) {
    // async-stage 64x32 f16 A tile and 64x32 f16 B tile (one b128 per lane each)
    async_ld16(lds_lo32(&lsA[i0]),
               (unsigned)(((m0 + rT) * 1024 + k0 + kT) * 2), Xh);
    async_ld16(lds_lo32(&lsB[i0]),
               (unsigned)(((n0 + rT) * 1024 + k0 + kT) * 2), Whz);
    if (k0 + 32 < 1024)   // gfx1250 global_prefetch_b8 for next K panel
      __builtin_prefetch(&Xh[(size_t)(m0 + (t >> 2)) * 1024 + k0 + 32], 0, 1);
    async_wait0();
    __syncthreads();

    v16h a, b0, b1;
#pragma unroll
    for (int j = 0; j < 8; ++j) {
      a[j]     = lsA[(mi * 16 + Ml) * 32 + kbA + j];
      a[8 + j] = lsA[(mi * 16 + Ml) * 32 + kbA + 8 + j];
    }
#pragma unroll
    for (int j = 0; j < 16; ++j) {
      b0[j] = lsB[((ni0    ) * 16 + Ml) * 32 + kbB + j];
      b1[j] = lsB[((ni0 + 1) * 16 + Ml) * 32 + kbB + j];
    }
    acc0 = WMMA_F16(a, b0, acc0);
    acc1 = WMMA_F16(a, b1, acc1);
    __syncthreads();
  }

  // epilogue: C layout -> [B,H,S,D]
  const int rbase = m0 + mi * 16 + ((lane < 16) ? 0 : 8);
#pragma unroll
  for (int j = 0; j < 8; ++j) {
    int row = rbase + j;
    int b = row >> 12, s = row & 4095;
    int col0 = n0 + ni0 * 16 + Ml;
    int col1 = col0 + 16;
    size_t o0 = (((size_t)b * 16 + (col0 >> 6)) * 4096 + s) * 64 + (col0 & 63);
    size_t o1 = (((size_t)b * 16 + (col1 >> 6)) * 4096 + s) * 64 + (col1 & 63);
    if (blockIdx.z == 0) {
      qkb[o0] = acc0[j];               qkb[o1] = acc1[j];
      qkh[o0] = (_Float16)acc0[j];     qkh[o1] = (_Float16)acc1[j];
    } else {
      vh[o0]  = (_Float16)acc0[j];     vh[o1]  = (_Float16)acc1[j];
    }
  }
}

// ------------------------------------------------------------------
// K2: LSH hashing. one thread per (b,h,s); rotations staged in LDS.
// ------------------------------------------------------------------
__global__ __launch_bounds__(256) void k_hash(const float* __restrict__ qkb,
                                              const float* __restrict__ rot,
                                              int* __restrict__ buckets) {
  __shared__ float lsRot[64 * 128];       // rot[d][n*64+r]
  const int t = threadIdx.x;
  for (int i = t; i < 8192; i += 256) lsRot[i] = rot[i];
  __syncthreads();

  const int gid = blockIdx.x * 256 + t;   // 0..131071
  const int bh = gid >> 12;
  const int s  = gid & 4095;
  const f4v* q4 = (const f4v*)(qkb + ((size_t)bh * 4096 + s) * 64);
  float qr[64];
#pragma unroll
  for (int p = 0; p < 16; ++p) {
    f4v x = q4[p];
    qr[4 * p] = x[0]; qr[4 * p + 1] = x[1]; qr[4 * p + 2] = x[2]; qr[4 * p + 3] = x[3];
  }

#pragma unroll 1
  for (int n = 0; n < 2; ++n) {
    float mx = -1e30f, mn = 1e30f;
    int amx = 0, amn = 0;
    for (int r = 0; r < 64; ++r) {
      float acc = 0.f;
#pragma unroll
      for (int d = 0; d < 64; ++d) acc += qr[d] * lsRot[d * 128 + n * 64 + r];
      if (acc > mx) { mx = acc; amx = r; }
      if (acc < mn) { mn = acc; amn = r; }
    }
    int bucket = (mx >= -mn) ? amx : (64 + amn);  // argmax over [rot, -rot]
    buckets[(size_t)bh * 8192 + n * 4096 + s] = bucket + n * 128;
  }
}

// ------------------------------------------------------------------
// K3: counting-sort support (256 bucket keys per (b,h))
// ------------------------------------------------------------------
__global__ void k_zero(int* __restrict__ p, int n) {
  int i = blockIdx.x * 256 + threadIdx.x;
  if (i < n) p[i] = 0;
}

__global__ void k_hist(const int* __restrict__ buckets, int* __restrict__ hist) {
  int i = blockIdx.x * 256 + threadIdx.x;  // 0..524287
  int bh = i >> 13;
  atomicAdd(&hist[bh * 256 + buckets[i]], 1);
}

__global__ __launch_bounds__(256) void k_sort(const int* __restrict__ buckets,
                                              const int* __restrict__ hist,
                                              int* __restrict__ sorted) {
  __shared__ int scan[256];
  __shared__ int soff[256];
  const int bh = blockIdx.x, t = threadIdx.x;
  int v = hist[bh * 256 + t];
  scan[t] = v;
  __syncthreads();
  for (int d = 1; d < 256; d <<= 1) {
    int x = (t >= d) ? scan[t - d] : 0;
    __syncthreads();
    scan[t] += x;
    __syncthreads();
  }
  soff[t] = scan[t] - v;                   // exclusive prefix
  __syncthreads();
  for (int c0 = 0; c0 < 8192; c0 += 256) { // in-order chunks ~ stability
    int j = c0 + t;
    int b = buckets[(size_t)bh * 8192 + j];
    int p = atomicAdd(&soff[b], 1);
    sorted[(size_t)bh * 8192 + p] = j;
    __syncthreads();
  }
}

// ------------------------------------------------------------------
// K5: chunked attention. one block per (chunk c, bh). 54KB LDS.
// Q staged via async-to-LDS from f16 qk; dots/probs share f32 LDS region.
// outputs scattered straight to per-hash-round buffers (undoes the sort).
// ------------------------------------------------------------------
__global__ __launch_bounds__(256) void k_attn(const float* __restrict__ qkb,
                                              const _Float16* __restrict__ qkh,
                                              const _Float16* __restrict__ vh,
                                              const int* __restrict__ sorted,
                                              float* __restrict__ gout,
                                              float* __restrict__ glog) {
  __shared__ __attribute__((aligned(16))) _Float16 lsK[128 * 64]; // normalized keys
  __shared__ __attribute__((aligned(16))) _Float16 lsV[64 * 128]; // V^T: [d][krow]
  __shared__ __attribute__((aligned(16))) float    lsDots[32 * 128];
  __shared__ __attribute__((aligned(16))) _Float16 lsQ[32 * 64];
  __shared__ int tickK[128];
  __shared__ int stQ[64];

  const int c  = blockIdx.x;             // 0..127 chunks
  const int bh = blockIdx.y;             // 0..31
  const int cp = (c + 127) & 127;        // previous chunk (rolled)
  const int t = threadIdx.x, lane = t & 31, w = t >> 5;
  const int* srt = sorted + (size_t)bh * 8192;
  const float* qbase = qkb + (size_t)bh * 4096 * 64;

  if (t < 64) stQ[t] = srt[c * 64 + t];
  if (t >= 64 && t < 192) {
    int r = t - 64;
    int ch = (r < 64) ? cp : c;          // kk = concat(prev, cur)
    tickK[r] = srt[ch * 64 + (r & 63)] & 4095;
  }
  __syncthreads();

  // stage normalized K (threads 0..127, f32 source) and V^T (threads 128..255)
  if (t < 128) {
    int r = t;
    const f4v* src = (const f4v*)(qbase + (size_t)tickK[r] * 64);
    f4v rv[16];
    float ss = 0.f;
#pragma unroll
    for (int p = 0; p < 16; ++p) {
      rv[p] = src[p];
      ss += rv[p][0] * rv[p][0] + rv[p][1] * rv[p][1] +
            rv[p][2] * rv[p][2] + rv[p][3] * rv[p][3];
    }
    float sc = rsqrtf(ss * (1.0f / 64.0f) + 1e-6f) * 0.125f;  // rms-norm / sqrt(D)
    h8* dst = (h8*)&lsK[r * 64];
#pragma unroll
    for (int p = 0; p < 8; ++p) {
      h8 o;
#pragma unroll
      for (int j2 = 0; j2 < 4; ++j2) {
        o[j2]     = (_Float16)(rv[2 * p][j2] * sc);
        o[4 + j2] = (_Float16)(rv[2 * p + 1][j2] * sc);
      }
      dst[p] = o;
    }
  } else {
    int r = t - 128;
    const h8* src = (const h8*)(vh + ((size_t)bh * 4096 + tickK[r]) * 64);
#pragma unroll
    for (int p = 0; p < 8; ++p) {
      h8 vv = src[p];
#pragma unroll
      for (int j2 = 0; j2 < 8; ++j2) lsV[(p * 8 + j2) * 128 + r] = vv[j2];
    }
  }
  __syncthreads();

  const int Ml  = lane & 15;
  const int kbA = (lane < 16) ? 0 : 16;
  const int kbB = (lane < 16) ? 0 : 16;

  for (int half = 0; half < 2; ++half) {
    const int qr0 = half * 32;
    // async-stage 32 gathered Q rows (f16), one b128 per lane
    {
      int i0 = t * 8;                    // 0..2040
      int r = i0 >> 6;                   // 0..31
      int d0 = i0 & 63;
      int pos = stQ[qr0 + r] & 4095;
      unsigned goff = (unsigned)((((unsigned)bh * 4096u + (unsigned)pos) * 64u + (unsigned)d0) * 2u);
      async_ld16(lds_lo32(&lsQ[i0]), goff, qkh);
      async_wait0();
    }
    __syncthreads();

    // GEMM1: dots[32x128] = Q[32x64] x K^T[64x128]
    {
      const int qi = w & 1, ki0 = (w >> 1) * 2;
      v8f acc0 = {}, acc1 = {};
#pragma unroll
      for (int kk = 0; kk < 64; kk += 32) {
        v16h a, b0, b1;
#pragma unroll
        for (int j = 0; j < 8; ++j) {
          a[j]     = lsQ[(qi * 16 + Ml) * 64 + kk + kbA + j];
          a[8 + j] = lsQ[(qi * 16 + Ml) * 64 + kk + kbA + 8 + j];
        }
#pragma unroll
        for (int j = 0; j < 16; ++j) {
          b0[j] = lsK[((ki0    ) * 16 + Ml) * 64 + kk + kbB + j];
          b1[j] = lsK[((ki0 + 1) * 16 + Ml) * 64 + kk + kbB + j];
        }
        acc0 = WMMA_F16(a, b0, acc0);
        acc1 = WMMA_F16(a, b1, acc1);
      }
      const int rl = qi * 16 + ((lane < 16) ? 0 : 8);
#pragma unroll
      for (int j = 0; j < 8; ++j) {
        lsDots[(rl + j) * 128 + (ki0    ) * 16 + Ml] = acc0[j];
        lsDots[(rl + j) * 128 + (ki0 + 1) * 16 + Ml] = acc1[j];
      }
    }
    __syncthreads();

    // masks + logsumexp + probs (one thread per q-row)
    if (t < 32) {
      int r = t;
      int tq = stQ[qr0 + r] & 4095;
      float mx = -1e30f;
      for (int k = 0; k < 128; ++k) {
        float dv = lsDots[r * 128 + k];
        int tk = tickK[k];
        dv = (tq == tk) ? -1e5f : ((tq >= tk) ? dv : -1e9f);
        lsDots[r * 128 + k] = dv;
        mx = fmaxf(mx, dv);
      }
      float sum = 0.f;
      for (int k = 0; k < 128; ++k) sum += __expf(lsDots[r * 128 + k] - mx);
      float logit = mx + __logf(sum);
      for (int k = 0; k < 128; ++k)
        lsDots[r * 128 + k] = __expf(lsDots[r * 128 + k] - logit);
      int idx = stQ[qr0 + r];
      glog[((size_t)bh * 2 + (idx >> 12)) * 4096 + (idx & 4095)] = logit;
    }
    __syncthreads();

    // GEMM2: out[32x64] = P[32x128] x V[128x64]
    {
      const int qi = w & 1, di = w >> 1;  // di 0..3
      v8f acc = {};
#pragma unroll
      for (int kk = 0; kk < 128; kk += 32) {
        v16h a, bf;
#pragma unroll
        for (int j = 0; j < 8; ++j) {
          a[j]     = (_Float16)lsDots[(qi * 16 + Ml) * 128 + kk + kbA + j];
          a[8 + j] = (_Float16)lsDots[(qi * 16 + Ml) * 128 + kk + kbA + 8 + j];
        }
#pragma unroll
        for (int j = 0; j < 16; ++j)
          bf[j] = lsV[(di * 16 + Ml) * 128 + kk + kbB + j];
        acc = WMMA_F16(a, bf, acc);
      }
      const int rl = qi * 16 + ((lane < 16) ? 0 : 8);
#pragma unroll
      for (int j = 0; j < 8; ++j) {
        int idx = stQ[qr0 + rl + j];
        int n = idx >> 12, pos = idx & 4095;
        gout[(((size_t)bh * 2 + n) * 4096 + pos) * 64 + di * 16 + Ml] = acc[j];
      }
    }
    __syncthreads();
  }
}

// ------------------------------------------------------------------
// K6: combine hash rounds with logsumexp weights -> [B, S, H*D]
// ------------------------------------------------------------------
__global__ __launch_bounds__(256) void k_combine(const float* __restrict__ gout,
                                                 const float* __restrict__ glog,
                                                 float* __restrict__ out) {
  size_t i = (size_t)blockIdx.x * 256 + threadIdx.x;  // (bh, s, d)
  int d = (int)(i & 63);
  int s = (int)((i >> 6) & 4095);
  int bh = (int)(i >> 18);
  int b = bh >> 4, h = bh & 15;
  float l0 = glog[((size_t)bh * 2 + 0) * 4096 + s];
  float l1 = glog[((size_t)bh * 2 + 1) * 4096 + s];
  float m = fmaxf(l0, l1);
  float w0 = __expf(l0 - m), w1 = __expf(l1 - m);
  float inv = 1.0f / (w0 + w1);
  float o0 = gout[(((size_t)bh * 2 + 0) * 4096 + s) * 64 + d];
  float o1 = gout[(((size_t)bh * 2 + 1) * 4096 + s) * 64 + d];
  out[(((size_t)b * 4096 + s) * 16 + h) * 64 + d] = (o0 * w0 + o1 * w1) * inv;
}

// ------------------------------------------------------------------
extern "C" void kernel_launch(void* const* d_in, const int* in_sizes, int n_in,
                              void* d_out, int out_size, void* d_ws, size_t ws_size,
                              hipStream_t stream) {
  const float* X   = (const float*)d_in[0];  // hidden_states [2,4096,1024]
  const float* Wqk = (const float*)d_in[1];  // [1024,1024]
  const float* Wv  = (const float*)d_in[2];  // [1024,1024]
  const float* rot = (const float*)d_in[3];  // [64,2,64]
  float* out = (float*)d_out;                // [2,4096,1024]

  // workspace carve (~154 MB)
  char* p = (char*)d_ws;
  _Float16* Xh   = (_Float16*)p; p += (size_t)8388608 * 2;   // X in f16
  _Float16* Wh   = (_Float16*)p; p += (size_t)2097152 * 2;   // [Wqk|Wv] f16
  float*    qkb  = (float*)p;    p += (size_t)8388608 * 4;   // qk f32 [B,H,S,D]
  _Float16* qkh  = (_Float16*)p; p += (size_t)8388608 * 2;   // qk f16
  _Float16* vh   = (_Float16*)p; p += (size_t)8388608 * 2;   // v  f16
  int*    buckets = (int*)p;     p += (size_t)524288 * 4;    // [BH, NH*S]
  int*    hist    = (int*)p;     p += (size_t)8192 * 4;      // [BH, 256]
  int*    sorted  = (int*)p;     p += (size_t)524288 * 4;
  float*  gout    = (float*)p;   p += (size_t)16777216 * 4;  // [BH, NH, S, D]
  float*  glog    = (float*)p;   p += (size_t)524288 * 4;    // [BH, NH, S]

  k_cvt    <<<4096, 256, 0, stream>>>(X,   Xh);
  k_cvt    <<<512,  256, 0, stream>>>(Wqk, Wh);
  k_cvt    <<<512,  256, 0, stream>>>(Wv,  Wh + 1048576);
  k_proj   <<<dim3(128, 16, 2), 256, 0, stream>>>(Xh, Wh, qkb, qkh, vh);
  k_hash   <<<512, 256, 0, stream>>>(qkb, rot, buckets);
  k_zero   <<<32, 256, 0, stream>>>(hist, 8192);
  k_hist   <<<2048, 256, 0, stream>>>(buckets, hist);
  k_sort   <<<32, 256, 0, stream>>>(buckets, hist, sorted);
  k_attn   <<<dim3(128, 32), 256, 0, stream>>>(qkb, qkh, vh, sorted, gout, glog);
  k_combine<<<32768, 256, 0, stream>>>(gout, glog, out);
}
